// Model_39676907881548
// MI455X (gfx1250) — matrix operand order repaired and checked by reading, not compiled, not run on hardware
//
#include <hip/hip_runtime.h>

// ---------------------------------------------------------------------------
// Flash-attention with full-tensor inverse scale, for MI455X (gfx1250).
//   y = softmax((Q K^T) / inv_scale) V      Q,K,V fp32; matmuls in bf16 WMMA
// Memory-bound on the 268MB inv_scale stream; bf16 WMMA (16x16x32) keeps the
// 3.4e10 matmul FLOPs off the critical path. K-tile staging uses the Tensor
// Data Mover (tensor_load_to_lds, 6-arg builtin form + s_wait_tensorcnt) with
// HW-applied LDS row padding; V-tile needs a transpose so it stays on the
// VGPR path.
// ---------------------------------------------------------------------------

typedef __attribute__((ext_vector_type(16))) __bf16 v16bf;
typedef __attribute__((ext_vector_type(8)))  float  v8f;
typedef __attribute__((ext_vector_type(4)))  unsigned int v4u;
typedef __attribute__((ext_vector_type(8)))  int v8i;
typedef __attribute__((ext_vector_type(4)))  int v4i;

constexpr int Bc = 16;     // batch * heads
constexpr int Sc = 2048;   // sequence length
constexpr int Dc = 128;    // head dim

constexpr int KSTR32 = 132;  // f32 elems per K-tile row (128 + 4 pad) == TDM pad cfg
constexpr int VSTR   = 40;   // bf16 elems per V^T-tile row (32 + 8 pad)
constexpr int PSTR   = 36;   // f32 elems per P-tile row (32 + 4 pad)
constexpr float LOG2E = 1.4426950408889634f;

// LDS layout (single block so the TDM descriptor's lds_addr is a known offset)
constexpr int K_OFF  = 0;                       // 32*132*4   = 16896 B
constexpr int VT_OFF = 32 * KSTR32 * 4;         // 128*40*2   = 10240 B
constexpr int P_OFF  = VT_OFF + Dc * VSTR * 2;  // 4*16*36*4  =  9216 B
constexpr int SMEM_BYTES = P_OFF + 4 * 16 * PSTR * 4;

#if defined(__HIP_DEVICE_COMPILE__) && __has_builtin(__builtin_amdgcn_tensor_load_to_lds)
#define HAVE_TDM 1
#else
#define HAVE_TDM 0
#endif

static __device__ __forceinline__ v8f zero8() {
  v8f z;
#pragma unroll
  for (int i = 0; i < 8; ++i) z[i] = 0.0f;
  return z;
}

// 16 contiguous fp32 (64B, 16B-aligned) -> bf16 A/B fragment half-K vector.
static __device__ __forceinline__ v16bf ld_f32x16_bf16(const float* p) {
  float4 a0 = ((const float4*)p)[0];
  float4 a1 = ((const float4*)p)[1];
  float4 a2 = ((const float4*)p)[2];
  float4 a3 = ((const float4*)p)[3];
  v16bf t;
  t[0]=(__bf16)a0.x;  t[1]=(__bf16)a0.y;  t[2]=(__bf16)a0.z;  t[3]=(__bf16)a0.w;
  t[4]=(__bf16)a1.x;  t[5]=(__bf16)a1.y;  t[6]=(__bf16)a1.z;  t[7]=(__bf16)a1.w;
  t[8]=(__bf16)a2.x;  t[9]=(__bf16)a2.y;  t[10]=(__bf16)a2.z; t[11]=(__bf16)a2.w;
  t[12]=(__bf16)a3.x; t[13]=(__bf16)a3.y; t[14]=(__bf16)a3.z; t[15]=(__bf16)a3.w;
  return t;
}

// 16 contiguous bf16 (32B, 16B-aligned) from LDS.
static __device__ __forceinline__ v16bf ld_bf16x16(const __bf16* p) {
  union { v16bf v; uint4 u[2]; } t;
  t.u[0] = ((const uint4*)p)[0];
  t.u[1] = ((const uint4*)p)[1];
  return t.v;
}

static __device__ __forceinline__ v8f wmma_bf16(v16bf a, v16bf b, v8f c) {
  // (neg_a, A, neg_b, B, c_mod, C, reuse_a, reuse_b)
  return __builtin_amdgcn_wmma_f32_16x16x32_bf16(false, a, false, b,
                                                 (short)0, c, false, false);
}

__global__ __launch_bounds__(128) void attn_fullscale_kernel(
    const float* __restrict__ Q, const float* __restrict__ K,
    const float* __restrict__ V, const float* __restrict__ ISF,
    float* __restrict__ O) {
  __shared__ __align__(16) unsigned char smem[SMEM_BYTES];
  float*  kf32 = (float*)(smem + K_OFF);    // K tile, fp32, key-major, 132/row
  __bf16* vt   = (__bf16*)(smem + VT_OFF);  // V tile, bf16, d-major (transposed)
  float*  pful = (float*)(smem + P_OFF);    // per-wave P relayout tiles

  const int tid  = threadIdx.x;          // 0..127
  const int wave = tid >> 5;             // 0..3
  const int lane = tid & 31;             // wave32
  const int qblocks = Sc / 64;
  const int b  = blockIdx.x / qblocks;
  const int q0 = (blockIdx.x % qblocks) * 64 + wave * 16;

  const int col   = lane & 15;           // N index / A row index
  const int rbase = (lane < 16) ? 0 : 8; // C/D row base

  // ---- Q tile (16 rows x 128) as four 16x32 bf16 A-fragments, loaded once ----
  v16bf qf[4];
  {
    const float* qrow = Q + ((size_t)b * Sc + q0 + col) * Dc;
#pragma unroll
    for (int f = 0; f < 4; ++f) {
      const float* p0 = qrow + f * 32 + ((lane < 16) ? 0 : 16);
      float4 a0 = *(const float4*)(p0);
      float4 a1 = *(const float4*)(p0 + 4);
      float4 a2 = *(const float4*)(p0 + 8);
      float4 a3 = *(const float4*)(p0 + 12);
      v16bf t;
      t[0]=(__bf16)a0.x;  t[1]=(__bf16)a0.y;  t[2]=(__bf16)a0.z;  t[3]=(__bf16)a0.w;
      t[4]=(__bf16)a1.x;  t[5]=(__bf16)a1.y;  t[6]=(__bf16)a1.z;  t[7]=(__bf16)a1.w;
      t[8]=(__bf16)a2.x;  t[9]=(__bf16)a2.y;  t[10]=(__bf16)a2.z; t[11]=(__bf16)a2.w;
      t[12]=(__bf16)a3.x; t[13]=(__bf16)a3.y; t[14]=(__bf16)a3.z; t[15]=(__bf16)a3.w;
      qf[f] = t;
    }
  }

  v8f acc_o[8];
#pragma unroll
  for (int t = 0; t < 8; ++t) acc_o[t] = zero8();
  float m_run[8], l_run[8];
#pragma unroll
  for (int g = 0; g < 8; ++g) { m_run[g] = -3.0e38f; l_run[g] = 0.0f; }

  float* pw = pful + wave * 16 * PSTR;

  for (int k0 = 0; k0 < Sc; k0 += 32) {
    // ---- K tile: Tensor Data Mover DMA (global -> padded LDS rows) ----
#if HAVE_TDM
    if (wave == 0) {
      unsigned long long ga =
          (unsigned long long)(const void*)(K + ((size_t)b * Sc + k0) * Dc);
      v4u g0;
      g0[0] = 1u;                                   // count=1 valid user D#
      g0[1] = (unsigned)K_OFF;                      // lds_addr (bytes)
      g0[2] = (unsigned)(ga & 0xFFFFFFFFu);         // global_addr[31:0]
      g0[3] = (unsigned)((ga >> 32) & 0x1FFFFFFu)   // global_addr[56:32]
              | (2u << 30);                         // type = 2 ("image")
      v8i g1;
      g1[0] = (2 << 16)     // data_size = 4B
            | (1 << 20)     // pad_enable
            | (6 << 22)     // pad_interval: 128 DWORDs
            | (3 << 25);    // pad_amount:   4 DWORDs  -> 132-float LDS rows
      g1[1] = (128 << 16);  // tensor_dim0 = 128 (bits 79:48, low half)
      g1[2] = (32 << 16);   // tensor_dim1 = 32  (bits 111:80, low half)
      g1[3] = (128 << 16);  // tile_dim0  = 128  (bits 127:112)
      g1[4] = 32;           // tile_dim1  = 32   (bits 143:128)
      g1[5] = 128;          // tensor_dim0_stride = 128 (bits 207:160, low)
      g1[6] = 0;
      g1[7] = 0;
      v4i g2 = {0, 0, 0, 0};            // 2D tensor: group2 unused
      v4i g3 = {0, 0, 0, 0};            // 2D tensor: group3 unused
      v8i g4 = {0, 0, 0, 0, 0, 0, 0, 0};// extra group in 6-arg builtin form
      __builtin_amdgcn_tensor_load_to_lds(g0, g1, g2, g3, g4, 0);
    }
#else
    {
      const float4* kg = (const float4*)(K + ((size_t)b * Sc + k0) * Dc);
#pragma unroll
      for (int i = 0; i < 8; ++i) {
        int j   = tid + i * 128;
        int key = (j * 4) >> 7;
        int d0  = (j * 4) & 127;
        *(float4*)&kf32[key * KSTR32 + d0] = kg[j];
      }
    }
#endif

    // ---- V tile: cooperative transpose-store (bf16, d-major) ----
    {
      const float4* vg = (const float4*)(V + ((size_t)b * Sc + k0) * Dc);
#pragma unroll
      for (int i = 0; i < 8; ++i) {
        int j   = tid + i * 128;
        int key = (j * 4) >> 7;
        int d0  = (j * 4) & 127;
        float4 v4 = vg[j];
        vt[(d0 + 0) * VSTR + key] = (__bf16)v4.x;
        vt[(d0 + 1) * VSTR + key] = (__bf16)v4.y;
        vt[(d0 + 2) * VSTR + key] = (__bf16)v4.z;
        vt[(d0 + 3) * VSTR + key] = (__bf16)v4.w;
      }
    }
#if HAVE_TDM
    __builtin_amdgcn_s_wait_tensorcnt(0);
#endif
    __syncthreads();

    // ---- S = Q K^T : 16 rows x 32 keys, 8 WMMAs (B frags read as fp32) ----
    v8f s0 = zero8(), s1 = zero8();
#pragma unroll
    for (int f = 0; f < 4; ++f) {
      const int dstart = f * 32 + ((lane < 16) ? 0 : 16);
      v16bf bk0 = ld_f32x16_bf16(&kf32[col * KSTR32 + dstart]);        // keys 0..15
      v16bf bk1 = ld_f32x16_bf16(&kf32[(col + 16) * KSTR32 + dstart]); // keys 16..31
      s0 = wmma_bf16(qf[f], bk0, s0);
      s1 = wmma_bf16(qf[f], bk1, s1);
    }

    // ---- divide by streamed inv_scale tensor, online softmax ----
    const float* scl = ISF + ((size_t)b * Sc + q0) * Sc + k0;
    if (k0 + 32 < Sc)
      __builtin_prefetch(scl + 32 + (size_t)(rbase + (lane & 7)) * Sc + col, 0, 3);

    float mc[8];
#pragma unroll
    for (int g = 0; g < 8; ++g) {
      float i0 = scl[(size_t)(rbase + g) * Sc + col];
      float i1 = scl[(size_t)(rbase + g) * Sc + col + 16];
      s0[g] = s0[g] / i0;
      s1[g] = s1[g] / i1;
      mc[g] = fmaxf(s0[g], s1[g]);
    }
#pragma unroll
    for (int off = 1; off < 16; off <<= 1) {
#pragma unroll
      for (int g = 0; g < 8; ++g)
        mc[g] = fmaxf(mc[g], __shfl_xor(mc[g], off, 32));
    }
    float sumc[8];
#pragma unroll
    for (int g = 0; g < 8; ++g) {
      float mnew  = fmaxf(m_run[g], mc[g]);
      float alpha = exp2f((m_run[g] - mnew) * LOG2E);
      m_run[g] = mnew;
      s0[g] = exp2f((s0[g] - mnew) * LOG2E);
      s1[g] = exp2f((s1[g] - mnew) * LOG2E);
      sumc[g] = s0[g] + s1[g];
      l_run[g] *= alpha;
#pragma unroll
      for (int t = 0; t < 8; ++t) acc_o[t][g] *= alpha;
    }
#pragma unroll
    for (int off = 1; off < 16; off <<= 1) {
#pragma unroll
      for (int g = 0; g < 8; ++g)
        sumc[g] += __shfl_xor(sumc[g], off, 32);
    }
#pragma unroll
    for (int g = 0; g < 8; ++g) l_run[g] += sumc[g];

    // ---- relayout P: C/D layout -> A layout via per-wave LDS tile ----
#pragma unroll
    for (int g = 0; g < 8; ++g) {
      pw[(rbase + g) * PSTR + col]      = s0[g];
      pw[(rbase + g) * PSTR + col + 16] = s1[g];
    }
    // Per-wave DS ops are executed in order (DScnt); just stop compiler motion.
    asm volatile("" ::: "memory");

    v16bf pa;
    {
      const float* pr = pw + col * PSTR + ((lane < 16) ? 0 : 16);   // row = lane%16, K base = 16*(lane/16)
      float4 f0 = *(const float4*)(pr);
      float4 f1 = *(const float4*)(pr + 4);
      float4 f2 = *(const float4*)(pr + 8);
      float4 f3 = *(const float4*)(pr + 12);
      pa[0]=(__bf16)f0.x;  pa[1]=(__bf16)f0.y;  pa[2]=(__bf16)f0.z;  pa[3]=(__bf16)f0.w;
      pa[4]=(__bf16)f1.x;  pa[5]=(__bf16)f1.y;  pa[6]=(__bf16)f1.z;  pa[7]=(__bf16)f1.w;
      pa[8]=(__bf16)f2.x;  pa[9]=(__bf16)f2.y;  pa[10]=(__bf16)f2.z; pa[11]=(__bf16)f2.w;
      pa[12]=(__bf16)f3.x; pa[13]=(__bf16)f3.y; pa[14]=(__bf16)f3.z; pa[15]=(__bf16)f3.w;
    }

    // ---- O += P V : 8 WMMAs over the 8 output-dim subtiles ----
#pragma unroll
    for (int t = 0; t < 8; ++t) {
      const __bf16* vp = &vt[(t * 16 + col) * VSTR + ((lane < 16) ? 0 : 16)];
      v16bf bv = ld_bf16x16(vp);
      acc_o[t] = wmma_bf16(pa, bv, acc_o[t]);
    }
    __syncthreads();
  }

  // ---- epilogue: O / l ----
  float* orow = O + ((size_t)b * Sc + q0) * Dc;
#pragma unroll
  for (int g = 0; g < 8; ++g) {
    float linv = 1.0f / l_run[g];
#pragma unroll
    for (int t = 0; t < 8; ++t)
      orow[(size_t)(rbase + g) * Dc + t * 16 + col] = acc_o[t][g] * linv;
  }
}

extern "C" void kernel_launch(void* const* d_in, const int* in_sizes, int n_in,
                              void* d_out, int out_size, void* d_ws, size_t ws_size,
                              hipStream_t stream) {
  (void)in_sizes; (void)n_in; (void)out_size; (void)d_ws; (void)ws_size;
  const float* Q   = (const float*)d_in[0];
  const float* K   = (const float*)d_in[1];
  const float* V   = (const float*)d_in[2];
  const float* ISF = (const float*)d_in[3];
  // d_in[4] = dropout_p (0) -> identity, ignored.
  float* O = (float*)d_out;

  dim3 grid(Bc * (Sc / 64));  // 512 workgroups: one per (batch, 64-row q block)
  dim3 block(128);            // 4 waves; each wave owns a 16-row q tile
  hipLaunchKernelGGL(attn_fullscale_kernel, grid, block, 0, stream,
                     Q, K, V, ISF, O);
}
